// MaskingDecoder_60455959658519
// MI455X (gfx1250) — matrix operand order repaired and checked by reading, not compiled, not run on hardware
//
#include <hip/hip_runtime.h>
#include <hip/hip_bf16.h>
#include <math.h>

// ---------------------------------------------------------------------------
// MI455X (gfx1250) implementation. wave32. All matmuls use
// v_wmma_f32_16x16x32_bf16 (bf16 inputs, fp32 accumulate).
// Weights are pre-converted to bf16 once so GEMM inner loops are pure
// load_b128 + v_wmma (no per-tile cvt VALU).
// ---------------------------------------------------------------------------

typedef __bf16 bf16;
typedef __attribute__((ext_vector_type(16))) __bf16 v16bf;
typedef __attribute__((ext_vector_type(8)))  __bf16 v8bf;
typedef __attribute__((ext_vector_type(8)))  float  v8f;

#define BB   4
#define NLAT 1024
#define NQRY 4096
#define DIMD 512
#define HDS  8
#define DHD  64

static __device__ __forceinline__ v8f wmma_bf(v16bf a, v16bf b, v8f c) {
  return __builtin_amdgcn_wmma_f32_16x16x32_bf16(false, a, false, b, (short)0, c, false, false);
}

// A fragment (16x32 bf16, M x K), row-major source. lane holds row = lane&15,
// half = lane>>4; elements 0..15 = K[16h..16h+15].
static __device__ __forceinline__ v16bf load_a_frag(const bf16* arow, int half) {
  v8bf lo = *reinterpret_cast<const v8bf*>(arow + 16 * half);
  v8bf hi = *reinterpret_cast<const v8bf*>(arow + 16 * half + 8);
  v16bf f;
#pragma unroll
  for (int t = 0; t < 8; ++t) { f[t] = lo[t]; f[8 + t] = hi[t]; }
  return f;
}

static __device__ __forceinline__ float block_reduce_sum(float v, float* sbuf) {
  int tid = threadIdx.x;
  sbuf[tid] = v; __syncthreads();
  for (int s = blockDim.x >> 1; s > 0; s >>= 1) {
    if (tid < s) sbuf[tid] += sbuf[tid + s];
    __syncthreads();
  }
  float r = sbuf[0]; __syncthreads();
  return r;
}

// inv_freq[i], i in [0,32): zero for i>=16 else 2pi/T_MIN * (T_MIN/T_MAX)^(i/16)
static __device__ __forceinline__ float inv_freq_dev(int i) {
  if (i >= 16) return 0.0f;
  const float T_MIN = 1e-4f, T_MAX = 2.0627f;
  return (6.28318530717958647692f / T_MIN) * powf(T_MIN / T_MAX, (float)i * (1.0f / 16.0f));
}

// ---------------------- fp32 -> bf16 weight conversion ---------------------
__global__ void k_cvt(const float* __restrict__ in, bf16* __restrict__ out, int n) {
  int i = blockIdx.x * 256 + threadIdx.x;
  if (i < n) out[i] = (bf16)in[i];
}

// ------------------------- qtok build + LayerNorm --------------------------
__global__ void __launch_bounds__(128) k_qtok_ln(
    const int* __restrict__ idx, const float* __restrict__ unit_emb,
    const float* __restrict__ vp_b, const float* __restrict__ w,
    const float* __restrict__ bb, float* __restrict__ qtok, bf16* __restrict__ xn) {
  __shared__ float sbuf[128];
  const int row = blockIdx.x;
  const float* ue = unit_emb + (size_t)idx[row] * 256;
  float v[4]; float s = 0.f, s2 = 0.f;
#pragma unroll
  for (int j = 0; j < 4; ++j) {
    int e = threadIdx.x + j * 128;
    float x = (e < 256) ? vp_b[e] : ue[e - 256];
    v[j] = x; s += x; s2 += x * x;
    qtok[(size_t)row * 512 + e] = x;
  }
  s  = block_reduce_sum(s, sbuf);
  s2 = block_reduce_sum(s2, sbuf);
  float mean = s * (1.f / 512.f);
  float rstd = rsqrtf(fmaxf(s2 * (1.f / 512.f) - mean * mean, 0.f) + 1e-5f);
#pragma unroll
  for (int j = 0; j < 4; ++j) {
    int e = threadIdx.x + j * 128;
    xn[(size_t)row * 512 + e] = (bf16)((v[j] - mean) * rstd * w[e] + bb[e]);
  }
}

// --------------------------- generic LN (D=512) ----------------------------
__global__ void __launch_bounds__(128) k_ln(
    const float* __restrict__ x, const float* __restrict__ w,
    const float* __restrict__ bb, bf16* __restrict__ out) {
  __shared__ float sbuf[128];
  const int row = blockIdx.x;
  const float* xr = x + (size_t)row * 512;
  float v[4]; float s = 0.f, s2 = 0.f;
#pragma unroll
  for (int j = 0; j < 4; ++j) {
    int e = threadIdx.x + j * 128;
    v[j] = xr[e]; s += v[j]; s2 += v[j] * v[j];
  }
  s  = block_reduce_sum(s, sbuf);
  s2 = block_reduce_sum(s2, sbuf);
  float mean = s * (1.f / 512.f);
  float rstd = rsqrtf(fmaxf(s2 * (1.f / 512.f) - mean * mean, 0.f) + 1e-5f);
#pragma unroll
  for (int j = 0; j < 4; ++j) {
    int e = threadIdx.x + j * 128;
    out[(size_t)row * 512 + e] = (bf16)((v[j] - mean) * rstd * w[e] + bb[e]);
  }
}

// ------------- GEMM: out[M,N] = A[M,K](bf16) @ W[N,K]^T (bf16) -------------
// optional bias[N] and residual[M,N].  block = 128 (4 waves), wave tile 32x64:
// each B fragment (one 32B v16bf load) feeds two WMMAs.
__global__ void __launch_bounds__(128) k_gemm(
    const bf16* __restrict__ A, const bf16* __restrict__ W,
    const float* __restrict__ bias, const float* __restrict__ residual,
    float* __restrict__ out, int M, int N, int K) {
  const int lane = threadIdx.x & 31, wave = threadIdx.x >> 5;
  const int n0 = blockIdx.x * 64;
  const int m0 = blockIdx.y * 128 + wave * 32;
  const int row = lane & 15, half = lane >> 4, col = lane & 15;
  v8f acc[2][4] = {};
  const bf16* ar0 = A + (size_t)(m0 + row) * K;
  const bf16* ar1 = A + (size_t)(m0 + 16 + row) * K;
  for (int k0 = 0; k0 < K; k0 += 32) {
    v16bf af0 = load_a_frag(ar0 + k0, half);
    v16bf af1 = load_a_frag(ar1 + k0, half);
#pragma unroll
    for (int nt = 0; nt < 4; ++nt) {
      v16bf bf = *reinterpret_cast<const v16bf*>(
          W + (size_t)(n0 + nt * 16 + col) * K + k0 + 16 * half);
      acc[0][nt] = wmma_bf(af0, bf, acc[0][nt]);
      acc[1][nt] = wmma_bf(af1, bf, acc[1][nt]);
    }
  }
#pragma unroll
  for (int sub = 0; sub < 2; ++sub)
#pragma unroll
    for (int nt = 0; nt < 4; ++nt) {
      int n = n0 + nt * 16 + col;
      float bv = bias ? bias[n] : 0.f;
#pragma unroll
      for (int r = 0; r < 8; ++r) {
        int m = m0 + sub * 16 + half * 8 + r;
        size_t o = (size_t)m * N + n;
        float v = acc[sub][nt][r] + bv;
        if (residual) v += residual[o];
        out[o] = v;
      }
    }
}

// -------- GEMM + gated exact-GELU epilogue: act = a * gelu(g), bf16 --------
// W (bf16) has 2*Nh rows: a-rows [0,Nh), g-rows [Nh,2Nh).  Wave tile 32x64.
__global__ void __launch_bounds__(128) k_gemm_gated(
    const bf16* __restrict__ A, const bf16* __restrict__ W,
    const float* __restrict__ bias, bf16* __restrict__ out, int M, int Nh, int K) {
  const int lane = threadIdx.x & 31, wave = threadIdx.x >> 5;
  const int n0 = blockIdx.x * 64;
  const int m0 = blockIdx.y * 128 + wave * 32;
  const int row = lane & 15, half = lane >> 4, col = lane & 15;
  v8f acca[2][4] = {}, accg[2][4] = {};
  const bf16* ar0 = A + (size_t)(m0 + row) * K;
  const bf16* ar1 = A + (size_t)(m0 + 16 + row) * K;
  for (int k0 = 0; k0 < K; k0 += 32) {
    v16bf af0 = load_a_frag(ar0 + k0, half);
    v16bf af1 = load_a_frag(ar1 + k0, half);
#pragma unroll
    for (int nt = 0; nt < 4; ++nt) {
      int n = n0 + nt * 16 + col;
      v16bf bfa = *reinterpret_cast<const v16bf*>(W + (size_t)n * K + k0 + 16 * half);
      v16bf bfg = *reinterpret_cast<const v16bf*>(W + (size_t)(Nh + n) * K + k0 + 16 * half);
      acca[0][nt] = wmma_bf(af0, bfa, acca[0][nt]);
      acca[1][nt] = wmma_bf(af1, bfa, acca[1][nt]);
      accg[0][nt] = wmma_bf(af0, bfg, accg[0][nt]);
      accg[1][nt] = wmma_bf(af1, bfg, accg[1][nt]);
    }
  }
#pragma unroll
  for (int sub = 0; sub < 2; ++sub)
#pragma unroll
    for (int nt = 0; nt < 4; ++nt) {
      int n = n0 + nt * 16 + col;
      float ba = bias[n], bg = bias[Nh + n];
#pragma unroll
      for (int r = 0; r < 8; ++r) {
        int m = m0 + sub * 16 + half * 8 + r;
        float a = acca[sub][nt][r] + ba;
        float g = accg[sub][nt][r] + bg;
        float ge = 0.5f * g * (1.0f + erff(g * 0.70710678118654752f));
        out[(size_t)m * Nh + n] = (bf16)(a * ge);
      }
    }
}

// ---------------------------- rotary kernels -------------------------------
__global__ void k_rot_q(const float* __restrict__ qraw, const float* __restrict__ ts,
                        bf16* __restrict__ qbuf) {
  int tid = blockIdx.x * blockDim.x + threadIdx.x;
  int pr = tid & 31; int tmp = tid >> 5;
  int h = tmp & (HDS - 1); tmp >>= 3;
  int q = tmp & (NQRY - 1); int b = tmp >> 12;
  float t = ts[b * NQRY + q];
  float f = t * inv_freq_dev(pr);
  float c = cosf(f), s = sinf(f);
  const float* src = qraw + (((size_t)b * NQRY + q) * HDS + h) * DHD + 2 * pr;
  float x0 = src[0], x1 = src[1];
  bf16* dst = qbuf + (((size_t)(b * HDS + h)) * NQRY + q) * DHD + 2 * pr;
  dst[0] = (bf16)(x0 * c - x1 * s);
  dst[1] = (bf16)(x1 * c + x0 * s);
}

__global__ void k_rot_kv(const float* __restrict__ kvraw, const float* __restrict__ ts,
                         bf16* __restrict__ kbuf, bf16* __restrict__ vtbuf) {
  int tid = blockIdx.x * blockDim.x + threadIdx.x;
  int pr = tid & 31; int tmp = tid >> 5;
  int h = tmp & (HDS - 1); tmp >>= 3;
  int l = tmp & (NLAT - 1); int b = tmp >> 10;
  float t = ts[b * NLAT + l];
  float f = t * inv_freq_dev(pr);
  float c = cosf(f), s = sinf(f);
  const float* kvr = kvraw + ((size_t)b * NLAT + l) * 1024;
  float k0 = kvr[h * DHD + 2 * pr],       k1 = kvr[h * DHD + 2 * pr + 1];
  float v0 = kvr[512 + h * DHD + 2 * pr], v1 = kvr[512 + h * DHD + 2 * pr + 1];
  bf16* kd = kbuf + (((size_t)(b * HDS + h)) * NLAT + l) * DHD + 2 * pr;
  kd[0] = (bf16)(k0 * c - k1 * s);
  kd[1] = (bf16)(k1 * c + k0 * s);
  // v stored transposed: (B,H,DH,NL)
  bf16* vd = vtbuf + ((size_t)(b * HDS + h) * DHD) * NLAT;
  vd[(size_t)(2 * pr)     * NLAT + l] = (bf16)(v0 * c - v1 * s);
  vd[(size_t)(2 * pr + 1) * NLAT + l] = (bf16)(v1 * c + v0 * s);
}

// reverse rotary on attention output, emit bf16 (B,NQ,H*DH)
__global__ void k_unrot(const float* __restrict__ ao, const float* __restrict__ ts,
                        bf16* __restrict__ out) {
  int tid = blockIdx.x * blockDim.x + threadIdx.x;
  int pr = tid & 31; int tmp = tid >> 5;
  int h = tmp & (HDS - 1); tmp >>= 3;
  int q = tmp & (NQRY - 1); int b = tmp >> 12;
  float t = ts[b * NQRY + q];
  float f = t * inv_freq_dev(pr);
  float c = cosf(f), s = sinf(f);
  const float* src = ao + (((size_t)b * NQRY + q) * HDS + h) * DHD + 2 * pr;
  float x0 = src[0], x1 = src[1];
  bf16* dst = out + ((size_t)b * NQRY + q) * 512 + h * DHD + 2 * pr;
  dst[0] = (bf16)(x0 * c + x1 * s);   // rot(-f): cos same, sin negated
  dst[1] = (bf16)(x1 * c - x0 * s);
}

// ------------------- flash-style WMMA cross-attention ----------------------
// wave per (b, h, 16-query tile); online softmax over NL=1024 in chunks of 64.
__global__ void __launch_bounds__(128) k_attn(
    const bf16* __restrict__ qb, const bf16* __restrict__ kb,
    const bf16* __restrict__ vt, float* __restrict__ ob) {
  __shared__ bf16 lds[4][16 * 64];          // per-wave P tile (row-major 16x64)
  const int lane = threadIdx.x & 31, wave = threadIdx.x >> 5;
  const int gid = blockIdx.x * 4 + wave;    // 0 .. B*H*(NQ/16)-1
  const int qtiles = NQRY / 16;
  const int qt = gid % qtiles;
  const int h  = (gid / qtiles) % HDS;
  const int b  = gid / (qtiles * HDS);
  const int row = lane & 15, half = lane >> 4, col = lane & 15;

  const bf16* qrow = qb + (((size_t)(b * HDS + h)) * NQRY + qt * 16 + row) * DHD;
  v16bf aq0 = load_a_frag(qrow, half);       // K(dh) 0..31
  v16bf aq1 = load_a_frag(qrow + 32, half);  // K(dh) 32..63

  const bf16* kbase = kb + (size_t)(b * HDS + h) * NLAT * DHD;
  const bf16* vbase = vt + (size_t)(b * HDS + h) * DHD * NLAT;

  v8f o[4] = {};
  float mr[8], lr[8];
#pragma unroll
  for (int r = 0; r < 8; ++r) { mr[r] = -1e30f; lr[r] = 0.f; }

  bf16* lp = lds[wave];
  for (int lc = 0; lc < NLAT; lc += 64) {
    // ---- scores tile S = Q(16x64) . K^T(64x64chunk), scaled by 1/8 ----
    v8f s[4] = {};
#pragma unroll
    for (int nt = 0; nt < 4; ++nt) {
      const bf16* kr = kbase + (size_t)(lc + nt * 16 + col) * DHD + 16 * half;
      v16bf bk0 = *reinterpret_cast<const v16bf*>(kr);
      v16bf bk1 = *reinterpret_cast<const v16bf*>(kr + 32);
      s[nt] = wmma_bf(aq0, bk0, s[nt]);
      s[nt] = wmma_bf(aq1, bk1, s[nt]);
    }
    // ---- online softmax update (rows live in 16-lane half-groups) ----
    float mnew[8], sc[8];
#pragma unroll
    for (int r = 0; r < 8; ++r) {
      float mx = fmaxf(fmaxf(s[0][r], s[1][r]), fmaxf(s[2][r], s[3][r])) * 0.125f;
      for (int d = 1; d < 16; d <<= 1) mx = fmaxf(mx, __shfl_xor(mx, d));
      mnew[r] = fmaxf(mr[r], mx);
      sc[r] = __expf(mr[r] - mnew[r]);
    }
    float p[4][8];
#pragma unroll
    for (int nt = 0; nt < 4; ++nt)
#pragma unroll
      for (int r = 0; r < 8; ++r)
        p[nt][r] = __expf(s[nt][r] * 0.125f - mnew[r]);
#pragma unroll
    for (int r = 0; r < 8; ++r) {
      float rs = p[0][r] + p[1][r] + p[2][r] + p[3][r];
      for (int d = 1; d < 16; d <<= 1) rs += __shfl_xor(rs, d);
      lr[r] = lr[r] * sc[r] + rs;
#pragma unroll
      for (int nt = 0; nt < 4; ++nt) o[nt][r] *= sc[r];
      mr[r] = mnew[r];
    }
    // ---- C-layout P -> row-major LDS -> A-layout fragments ----
#pragma unroll
    for (int nt = 0; nt < 4; ++nt)
#pragma unroll
      for (int r = 0; r < 8; ++r)
        lp[(half * 8 + r) * 64 + nt * 16 + col] = (bf16)p[nt][r];
    __builtin_amdgcn_wave_barrier();
    asm volatile("s_wait_dscnt 0" ::: "memory");   // DS ops in-order per wave
    v16bf ap0 = load_a_frag(lp + row * 64, half);       // K(latent) 0..31
    v16bf ap1 = load_a_frag(lp + row * 64 + 32, half);  // K(latent) 32..63
    __builtin_amdgcn_wave_barrier();
    // ---- O += P(16x64chunk) . V(64chunk x 64dh), V stored dh-major ----
#pragma unroll
    for (int nt = 0; nt < 4; ++nt) {
      const bf16* vr = vbase + (size_t)(nt * 16 + col) * NLAT + lc + 16 * half;
      v16bf bv0 = *reinterpret_cast<const v16bf*>(vr);
      v16bf bv1 = *reinterpret_cast<const v16bf*>(vr + 32);
      o[nt] = wmma_bf(ap0, bv0, o[nt]);
      o[nt] = wmma_bf(ap1, bv1, o[nt]);
    }
  }
  // ---- normalize and store (B, NQ, H, DH) fp32 ----
#pragma unroll
  for (int r = 0; r < 8; ++r) {
    float inv = 1.0f / lr[r];
    int m = qt * 16 + half * 8 + r;
#pragma unroll
    for (int nt = 0; nt < 4; ++nt)
      ob[(((size_t)b * NQRY + m) * HDS + h) * DHD + nt * 16 + col] = o[nt][r] * inv;
  }
}

// ------------- final: LN(dec2) . wproj + bproj, one row/block --------------
__global__ void __launch_bounds__(128) k_final(
    const float* __restrict__ x, const float* __restrict__ w,
    const float* __restrict__ bb, const float* __restrict__ wproj,
    const float* __restrict__ bproj, float* __restrict__ out) {
  __shared__ float sbuf[128];
  const int row = blockIdx.x;
  const float* xr = x + (size_t)row * 512;
  float v[4]; float s = 0.f, s2 = 0.f;
#pragma unroll
  for (int j = 0; j < 4; ++j) {
    int e = threadIdx.x + j * 128;
    v[j] = xr[e]; s += v[j]; s2 += v[j] * v[j];
  }
  s  = block_reduce_sum(s, sbuf);
  s2 = block_reduce_sum(s2, sbuf);
  float mean = s * (1.f / 512.f);
  float rstd = rsqrtf(fmaxf(s2 * (1.f / 512.f) - mean * mean, 0.f) + 1e-5f);
  float d = 0.f;
#pragma unroll
  for (int j = 0; j < 4; ++j) {
    int e = threadIdx.x + j * 128;
    d += ((v[j] - mean) * rstd * w[e] + bb[e]) * wproj[e];
  }
  d = block_reduce_sum(d, sbuf);
  if (threadIdx.x == 0) out[row] = d + bproj[0];
}

// ---------------------------------------------------------------------------
extern "C" void kernel_launch(void* const* d_in, const int* in_sizes, int n_in,
                              void* d_out, int out_size, void* d_ws, size_t ws_size,
                              hipStream_t stream) {
  (void)in_sizes; (void)n_in; (void)out_size; (void)ws_size;
  const float* enc       = (const float*)d_in[0];
  const float* lts       = (const float*)d_in[1];
  const int*   qidx      = (const int*)  d_in[2];
  const float* qts       = (const float*)d_in[3];
  const float* unit_emb  = (const float*)d_in[4];
  /* d_in[5] = vp_w, multiplied by 0.0 in reference */
  const float* vp_b      = (const float*)d_in[6];
  const float* ca_norm_w = (const float*)d_in[7];
  const float* ca_norm_b = (const float*)d_in[8];
  const float* ca_ctx_w  = (const float*)d_in[9];
  const float* ca_ctx_b  = (const float*)d_in[10];
  const float* wq        = (const float*)d_in[11];
  const float* wkv       = (const float*)d_in[12];
  const float* wo        = (const float*)d_in[13];
  const float* bo        = (const float*)d_in[14];
  const float* ffn_ln_w  = (const float*)d_in[15];
  const float* ffn_ln_b  = (const float*)d_in[16];
  const float* w1        = (const float*)d_in[17];
  const float* b1        = (const float*)d_in[18];
  const float* w2        = (const float*)d_in[19];
  const float* b2        = (const float*)d_in[20];
  const float* out_ln_w  = (const float*)d_in[21];
  const float* out_ln_b  = (const float*)d_in[22];
  const float* wproj     = (const float*)d_in[23];
  const float* bproj     = (const float*)d_in[24];

  const size_t MQ = (size_t)BB * NQRY;   // 16384 query rows
  const size_t ML = (size_t)BB * NLAT;   // 4096 latent rows

  char* ws = (char*)d_ws;
  size_t off = 0;
  auto alloc = [&](size_t bytes) -> void* {
    void* p = ws + off;
    off += (bytes + 255) & ~(size_t)255;
    return p;
  };
  float* qtok  = (float*)alloc(MQ * 512 * 4);   // residual source
  bf16*  xn    = (bf16*) alloc(MQ * 512 * 2);   // LN(qtok); reused as hln
  bf16*  cn    = (bf16*) alloc(ML * 512 * 2);   // LN(latents)
  float* qraw  = (float*)alloc(MQ * 512 * 4);   // reused as attn output
  float* kvraw = (float*)alloc(ML * 1024 * 4);  // reused as ao_bf16
  bf16*  qbuf  = (bf16*) alloc(MQ * 512 * 2);   // (B,H,NQ,DH)
  bf16*  kbuf  = (bf16*) alloc(ML * 512 * 2);   // (B,H,NL,DH)
  bf16*  vtbuf = (bf16*) alloc(ML * 512 * 2);   // (B,H,DH,NL)
  float* dec   = (float*)alloc(MQ * 512 * 4);
  bf16*  act   = (bf16*) alloc(MQ * 2048 * 2);  // gated FFN activation
  float* dec2  = (float*)alloc(MQ * 512 * 4);
  // bf16 copies of all GEMM weights (converted once, L2-resident: ~8 MB)
  bf16*  wqb   = (bf16*) alloc((size_t)512 * 512 * 2);
  bf16*  wkvb  = (bf16*) alloc((size_t)1024 * 512 * 2);
  bf16*  wob   = (bf16*) alloc((size_t)512 * 512 * 2);
  bf16*  w1b   = (bf16*) alloc((size_t)4096 * 512 * 2);
  bf16*  w2b   = (bf16*) alloc((size_t)512 * 2048 * 2);

  float* attnout = qraw;                        // alias (qraw dead after rotary)
  bf16*  ao      = (bf16*)kvraw;                // alias (kvraw dead after rotary)
  bf16*  hln     = xn;                          // alias (xn dead after q GEMM)

  dim3 blk(128);

  // ---- one-time weight conversion (tiny; overlaps nothing critical) ----
  k_cvt<<<(512 * 512 + 255) / 256,  256, 0, stream>>>(wq,  wqb,  512 * 512);
  k_cvt<<<(1024 * 512 + 255) / 256, 256, 0, stream>>>(wkv, wkvb, 1024 * 512);
  k_cvt<<<(512 * 512 + 255) / 256,  256, 0, stream>>>(wo,  wob,  512 * 512);
  k_cvt<<<(4096 * 512 + 255) / 256, 256, 0, stream>>>(w1,  w1b,  4096 * 512);
  k_cvt<<<(512 * 2048 + 255) / 256, 256, 0, stream>>>(w2,  w2b,  512 * 2048);

  k_qtok_ln<<<(int)MQ, blk, 0, stream>>>(qidx, unit_emb, vp_b, ca_norm_w, ca_norm_b, qtok, xn);
  k_ln     <<<(int)ML, blk, 0, stream>>>(enc, ca_ctx_w, ca_ctx_b, cn);

  // q = xn @ wq^T   (16384 x 512 x 512)
  k_gemm<<<dim3(512 / 64, (int)(MQ / 128)), blk, 0, stream>>>(xn, wqb, nullptr, nullptr, qraw, (int)MQ, 512, 512);
  // kv = cn @ wkv^T (4096 x 1024 x 512)
  k_gemm<<<dim3(1024 / 64, (int)(ML / 128)), blk, 0, stream>>>(cn, wkvb, nullptr, nullptr, kvraw, (int)ML, 1024, 512);

  const int nq_pairs = BB * NQRY * HDS * 32;    // 4,194,304
  const int nl_pairs = BB * NLAT * HDS * 32;    // 1,048,576
  k_rot_q <<<nq_pairs / 256, 256, 0, stream>>>(qraw, qts, qbuf);
  k_rot_kv<<<nl_pairs / 256, 256, 0, stream>>>(kvraw, lts, kbuf, vtbuf);

  // flash attention: B*H*(NQ/16) waves, 4 waves per block
  k_attn<<<(BB * HDS * (NQRY / 16)) / 4, blk, 0, stream>>>(qbuf, kbuf, vtbuf, attnout);

  k_unrot<<<nq_pairs / 256, 256, 0, stream>>>(attnout, qts, ao);

  // dec = qtok + ao @ wo^T + bo
  k_gemm<<<dim3(512 / 64, (int)(MQ / 128)), blk, 0, stream>>>(ao, wob, bo, qtok, dec, (int)MQ, 512, 512);

  // act = a * gelu(g), h = LN(dec) @ w1^T + b1
  k_ln<<<(int)MQ, blk, 0, stream>>>(dec, ffn_ln_w, ffn_ln_b, hln);
  k_gemm_gated<<<dim3(2048 / 64, (int)(MQ / 128)), blk, 0, stream>>>(hln, w1b, b1, act, (int)MQ, 2048, 512);

  // dec2 = dec + act @ w2^T + b2   (K = 2048)
  k_gemm<<<dim3(512 / 64, (int)(MQ / 128)), blk, 0, stream>>>(act, w2b, b2, dec, dec2, (int)MQ, 512, 2048);

  // out = LN(dec2) @ wproj^T + bproj
  k_final<<<(int)MQ, blk, 0, stream>>>(dec2, out_ln_w, out_ln_b, wproj, bproj, (float*)d_out);
}
